// SimplePerformer_52913997086812
// MI455X (gfx1250) — matrix operand order repaired and checked by reading, not compiled, not run on hardware
//
#include <hip/hip_runtime.h>

#define BB   8
#define GG   20000
#define DDIM 256
#define HH   8
#define DH   32
#define LLAY 4
#define MFEA 128
#define FF   1024
#define BG   (BB * GG)

typedef __attribute__((ext_vector_type(16))) __bf16 v16bf;
typedef __attribute__((ext_vector_type(8)))  float  v8f;
typedef __attribute__((ext_vector_type(4)))  unsigned int v4u;
typedef __attribute__((ext_vector_type(8)))  int v8i;
typedef __attribute__((ext_vector_type(4)))  int v4i;

union Frag16 { v16bf v; unsigned int u[8]; };

__device__ __forceinline__ unsigned short f2bf(float f) {
  unsigned int u = __float_as_uint(f);
  u += 0x7FFFu + ((u >> 16) & 1u);          // round-to-nearest-even
  return (unsigned short)(u >> 16);
}
__device__ __forceinline__ float bf2f(unsigned short h) {
  return __uint_as_float(((unsigned int)h) << 16);
}

// ---------------------------------------------------------------------------
// Generic WMMA GEMM: C[M,N] = act(A[M,K] @ W[K,N] + bias) + res1 + res2
// 64x128 block tile, 8 waves; each wave 16 rows x 64 cols = 4 WMMAs per
// k-step reusing a single A fragment (raises math per LDS byte / barrier).
// ACT: 0 = none, 1 = relu, 2 = exact gelu.
// ---------------------------------------------------------------------------
template <int ACT>
__global__ __launch_bounds__(256) void gemm_bf16_wmma(
    const float* __restrict__ A, const float* __restrict__ W,
    const float* __restrict__ bias, const float* __restrict__ res1,
    const float* __restrict__ res2, float* __restrict__ C, int N, int K) {
  __shared__ unsigned short As[64][34];    // [row][k]  bf16
  __shared__ unsigned short Ws[128][34];   // [col][k]  bf16 (W transposed)
  const int tid  = threadIdx.x;
  const int wave = tid >> 5, lane = tid & 31;
  const int half = lane >> 4, l16 = lane & 15;
  const int row0 = blockIdx.x * 64;
  const int col0 = blockIdx.y * 128;
  const int wrow = (wave & 3) * 16;
  const int wcol = (wave >> 2) * 64;
  v8f acc[4];
#pragma unroll
  for (int j = 0; j < 4; ++j) acc[j] = (v8f){0.f, 0.f, 0.f, 0.f, 0.f, 0.f, 0.f, 0.f};
  for (int kk = 0; kk < K; kk += 32) {
#pragma unroll
    for (int i = 0; i < 8; ++i) {
      int e = tid + i * 256;
      int r = e >> 5, c = e & 31;
      As[r][c] = f2bf(A[(size_t)(row0 + r) * K + kk + c]);
    }
#pragma unroll
    for (int i = 0; i < 16; ++i) {
      int e = tid + i * 256;
      int r = e >> 5, c = e & 31;
      Ws[r][c] = f2bf(W[(size_t)(kk + c) * N + col0 + r]);
    }
    if (kk + 32 < K)  // gfx1250 global_prefetch of next A tile
      __builtin_prefetch(A + (size_t)(row0 + (tid >> 2)) * K + kk + 32, 0, 1);
    __syncthreads();
    Frag16 a;
#pragma unroll
    for (int p = 0; p < 8; ++p) {
      int kb = 16 * half + 2 * p;
      a.u[p] = *(const unsigned int*)&As[wrow + l16][kb];
    }
#pragma unroll
    for (int j = 0; j < 4; ++j) {
      Frag16 b;
#pragma unroll
      for (int p = 0; p < 8; ++p)
        b.u[p] = *(const unsigned int*)&Ws[wcol + 16 * j + l16][16 * half + 2 * p];
      acc[j] = __builtin_amdgcn_wmma_f32_16x16x32_bf16(false, a.v, false, b.v,
                                                       (short)0, acc[j], false, false);
    }
    __syncthreads();
  }
#pragma unroll
  for (int j = 0; j < 4; ++j) {
#pragma unroll
    for (int r = 0; r < 8; ++r) {
      int gm = row0 + wrow + 8 * half + r;
      int gn = col0 + wcol + 16 * j + l16;
      float v0 = acc[j][r];
      if (bias) v0 += bias[gn];
      if (ACT == 1) v0 = fmaxf(v0, 0.f);
      if (ACT == 2) v0 = 0.5f * v0 * (1.f + erff(v0 * 0.70710678118f));
      if (res1) v0 += res1[(size_t)gm * N + gn];
      if (res2) v0 += res2[(size_t)gm * N + gn];
      C[(size_t)gm * N + gn] = v0;
    }
  }
}

// ---------------------------------------------------------------------------
// FAVOR+ softmax-kernel features. One block = 16 tokens x all 128 features
// for one (b,h). dd = (x*dn) @ proj^T via 8 WMMAs (K = DH = 32).
// MODE 0: queries (per-row max, write qp bf16)
// MODE 1: keys pass 1 (write per-row max only)
// MODE 2: keys pass 2 (use global kmax, write kp bf16)
// ---------------------------------------------------------------------------
template <int MODE>
__global__ __launch_bounds__(256) void feat_kernel(
    const float* __restrict__ QK, const float* __restrict__ proj,
    unsigned short* __restrict__ outp, float* __restrict__ rowmax_out,
    const float* __restrict__ kmax_in) {
  __shared__ unsigned short Qs[16][34];
  __shared__ unsigned short Ps[128][34];
  __shared__ float dd[16][128];
  __shared__ float sdiag[16];
  __shared__ float srmax[16];
  const int tid  = threadIdx.x;
  const int wave = tid >> 5, lane = tid & 31;
  const int half = lane >> 4, l16 = lane & 15;
  const int bh = blockIdx.y, b = bh >> 3, h = bh & 7;
  const int g0 = blockIdx.x * 16;
  const float dn = 0.42044820762685725f;  // 32^-0.25
  for (int i = tid; i < 16 * 32; i += 256) {
    int r = i >> 5, c = i & 31;
    Qs[r][c] = f2bf(QK[((size_t)b * GG + g0 + r) * DDIM + h * DH + c] * dn);
  }
  for (int i = tid; i < 128 * 32; i += 256) {
    int r = i >> 5, c = i & 31;
    Ps[r][c] = f2bf(proj[(size_t)r * DH + c]);
  }
  __syncthreads();
  Frag16 a, bfr;
#pragma unroll
  for (int p = 0; p < 8; ++p) {
    int kb = 16 * half + 2 * p;
    a.u[p]   = *(const unsigned int*)&Qs[l16][kb];
    bfr.u[p] = *(const unsigned int*)&Ps[16 * wave + l16][16 * half + 2 * p];
  }
  v8f acc = {0.f, 0.f, 0.f, 0.f, 0.f, 0.f, 0.f, 0.f};
  acc = __builtin_amdgcn_wmma_f32_16x16x32_bf16(false, a.v, false, bfr.v,
                                                (short)0, acc, false, false);
#pragma unroll
  for (int r = 0; r < 8; ++r) dd[8 * half + r][16 * wave + l16] = acc[r];
  __syncthreads();
  if (tid < 16) {
    float s = 0.f;
    for (int c = 0; c < 32; ++c) { float q = bf2f(Qs[tid][c]); s += q * q; }
    sdiag[tid] = 0.5f * s;
    if (MODE != 2) {
      float mx = -3.4e38f;
      for (int j = 0; j < 128; ++j) mx = fmaxf(mx, dd[tid][j]);
      if (MODE == 0) srmax[tid] = mx;
      else rowmax_out[(size_t)bh * GG + g0 + tid] = mx;
    }
  }
  __syncthreads();
  if (MODE == 1) return;
  const float ratio = 0.0883883476483184f;  // 128^-0.5
  float km = (MODE == 2) ? kmax_in[bh] : 0.f;
  for (int i = tid; i < 16 * 128; i += 256) {
    int r = i >> 7, j = i & 127;
    float mval = (MODE == 0) ? srmax[r] : km;
    float e = ratio * (expf(dd[r][j] - sdiag[r] - mval) + 1e-4f);
    outp[((size_t)bh * GG + g0 + r) * MFEA + j] = f2bf(e);
  }
}

// kmax[bh] = max over g of rowmax[bh, g]
__global__ __launch_bounds__(256) void reduce_max_kernel(
    const float* __restrict__ rowmax, float* __restrict__ kmax) {
  __shared__ float sm[256];
  const int bh = blockIdx.x;
  float mx = -3.4e38f;
  for (int g = threadIdx.x; g < GG; g += 256)
    mx = fmaxf(mx, rowmax[(size_t)bh * GG + g]);
  sm[threadIdx.x] = mx;
  __syncthreads();
  for (int s = 128; s > 0; s >>= 1) {
    if (threadIdx.x < s) sm[threadIdx.x] = fmaxf(sm[threadIdx.x], sm[threadIdx.x + s]);
    __syncthreads();
  }
  if (threadIdx.x == 0) kmax[bh] = sm[0];
}

// ksum[bh, m] = sum over g of kp[bh, g, m]   (blockDim = 128)
__global__ __launch_bounds__(128) void ksum_kernel(
    const unsigned short* __restrict__ kp, float* __restrict__ ksum) {
  const int bh = blockIdx.x, m = threadIdx.x;
  float s = 0.f;
  for (int g = 0; g < GG; ++g) s += bf2f(kp[((size_t)bh * GG + g) * MFEA + m]);
  ksum[bh * MFEA + m] = s;
}

// ctx[bh, m, d] = sum_g kp[bh,g,m] * v[b,g,h*DH+d] — WMMA, K = G.
__global__ __launch_bounds__(256) void ctx_kernel(
    const unsigned short* __restrict__ kp, const float* __restrict__ v,
    float* __restrict__ ctx) {
  __shared__ unsigned short At[128][34];  // [m][g-chunk]
  __shared__ unsigned short Vt[32][34];   // [d][g-chunk]
  const int tid  = threadIdx.x;
  const int wave = tid >> 5, lane = tid & 31;
  const int half = lane >> 4, l16 = lane & 15;
  const int bh = blockIdx.x, b = bh >> 3, h = bh & 7;
  v8f acc0 = {0.f, 0.f, 0.f, 0.f, 0.f, 0.f, 0.f, 0.f};
  v8f acc1 = acc0;
  for (int g0 = 0; g0 < GG; g0 += 32) {
    for (int i = tid; i < 32 * 128; i += 256) {
      int gg = i >> 7, m = i & 127;
      At[m][gg] = kp[((size_t)bh * GG + g0 + gg) * MFEA + m];
    }
    for (int i = tid; i < 32 * 32; i += 256) {
      int gg = i >> 5, d = i & 31;
      Vt[d][gg] = f2bf(v[((size_t)b * GG + g0 + gg) * DDIM + h * DH + d]);
    }
    __syncthreads();
    Frag16 a, b0, b1;
#pragma unroll
    for (int p = 0; p < 8; ++p) {
      int kb = 16 * half + 2 * p;
      a.u[p]  = *(const unsigned int*)&At[16 * wave + l16][kb];
      b0.u[p] = *(const unsigned int*)&Vt[l16][16 * half + 2 * p];
      b1.u[p] = *(const unsigned int*)&Vt[16 + l16][16 * half + 2 * p];
    }
    acc0 = __builtin_amdgcn_wmma_f32_16x16x32_bf16(false, a.v, false, b0.v,
                                                   (short)0, acc0, false, false);
    acc1 = __builtin_amdgcn_wmma_f32_16x16x32_bf16(false, a.v, false, b1.v,
                                                   (short)0, acc1, false, false);
    __syncthreads();
  }
#pragma unroll
  for (int r = 0; r < 8; ++r) {
    int m = 16 * wave + 8 * half + r;
    ctx[((size_t)bh * MFEA + m) * DH + l16]      = acc0[r];
    ctx[((size_t)bh * MFEA + m) * DH + 16 + l16] = acc1[r];
  }
}

// o[b,g,h*DH+d] = dinv[bh,g] * sum_m qp[bh,g,m] * ctx[bh,m,d] — WMMA, K = 128.
// Qs tile (64 tokens x 32 features, bf16, stride 256B in memory) is staged
// into LDS by the Tensor Data Mover: wave 0 issues tensor_load_to_lds with a
// hand-packed 2-D D# descriptor, waits on TENSORcnt, then the workgroup
// barrier releases all waves.
__global__ __launch_bounds__(256) void attnout_kernel(
    const unsigned short* __restrict__ qp, const float* __restrict__ ctx,
    const float* __restrict__ dinv, float* __restrict__ o) {
  __shared__ unsigned short Qs[64][32];    // [token][m-chunk] (TDM target)
  __shared__ unsigned short Ct[32][130];   // [d][m]  (ctx transposed)
  const int tid  = threadIdx.x;
  const int wave = tid >> 5, lane = tid & 31;
  const int half = lane >> 4, l16 = lane & 15;
  const int bh = blockIdx.y, b = bh >> 3, h = bh & 7;
  const int g0 = blockIdx.x * 64;
  const int wrow = (wave & 3) * 16;
  const int ntile = wave >> 2;             // 0 or 1: d offset 16*ntile
  for (int i = tid; i < 128 * 32; i += 256) {
    int m = i >> 5, d = i & 31;
    Ct[d][m] = f2bf(ctx[((size_t)bh * MFEA + m) * DH + d]);
  }
  v8f acc = {0.f, 0.f, 0.f, 0.f, 0.f, 0.f, 0.f, 0.f};
  for (int kk = 0; kk < MFEA; kk += 32) {
    if (wave == 0) {
      // Tensor DMA: 2-D tile, 64 rows x 32 elems of 2 bytes, row stride 128.
      unsigned ldsa = (unsigned)(size_t)&Qs[0][0];   // flat addr low 32 = LDS addr
      unsigned long long ga =
          (unsigned long long)(size_t)(qp + ((size_t)bh * GG + g0) * MFEA + kk);
      v4u g0d = {1u,                                  // count=1, user descriptor
                 ldsa,                                // lds_addr
                 (unsigned)ga,                        // global_addr[31:0]
                 ((unsigned)(ga >> 32) & 0x01FFFFFFu) | 0x80000000u}; // type=2
      v8i g1d = {0x00010000,  // data_size = 2 bytes
                 0,           // atomic barrier addr = 0
                 0x00000010,  // tensor_dim0 = 1<<20 (hi bits), dim1 lo = 0
                 0x00200010,  // tensor_dim1 = 1<<20 (hi), tile_dim0 = 32
                 64,          // tile_dim1 = 64
                 128,         // tensor_dim0_stride = 128 elements
                 0, 0};
      v4i zz = {0, 0, 0, 0};
      v8i zz8 = {0, 0, 0, 0, 0, 0, 0, 0};
      __builtin_amdgcn_tensor_load_to_lds(g0d, g1d, zz, zz, zz8, 0);
      __builtin_amdgcn_s_wait_tensorcnt(0);
    }
    __syncthreads();
    Frag16 a, bfr;
#pragma unroll
    for (int p = 0; p < 8; ++p) {
      int kb = 16 * half + 2 * p;
      a.u[p]   = *(const unsigned int*)&Qs[wrow + l16][kb];
      bfr.u[p] = *(const unsigned int*)&Ct[16 * ntile + l16][kk + 16 * half + 2 * p];
    }
    acc = __builtin_amdgcn_wmma_f32_16x16x32_bf16(false, a.v, false, bfr.v,
                                                  (short)0, acc, false, false);
    __syncthreads();
  }
#pragma unroll
  for (int r = 0; r < 8; ++r) {
    int g = g0 + wrow + 8 * half + r;
    if (g < GG) {
      float di = dinv[(size_t)bh * GG + g];
      o[((size_t)b * GG + g) * DDIM + h * DH + 16 * ntile + l16] = acc[r] * di;
    }
  }
}

// dinv[bh,g] = 1 / dot(qp[bh,g,:], ksum[bh,:])
__global__ __launch_bounds__(256) void dinv_kernel(
    const unsigned short* __restrict__ qp, const float* __restrict__ ksum,
    float* __restrict__ dinv) {
  const int g = blockIdx.x * 256 + threadIdx.x;
  const int bh = blockIdx.y;
  if (g >= GG) return;
  const unsigned short* row = qp + ((size_t)bh * GG + g) * MFEA;
  const float* ks = ksum + bh * MFEA;
  float s = 0.f;
  for (int j = 0; j < MFEA; ++j) s += bf2f(row[j]) * ks[j];
  dinv[(size_t)bh * GG + g] = 1.0f / s;
}

// LayerNorm over D=256: one wave32 per token, 8 tokens per block.
__global__ __launch_bounds__(256) void ln_kernel(
    const float* __restrict__ x, const float* __restrict__ gamma,
    const float* __restrict__ beta, float* __restrict__ y) {
  const int token = blockIdx.x * 8 + (threadIdx.x >> 5);
  const int lane  = threadIdx.x & 31;
  const float* row = x + (size_t)token * DDIM;
  float vals[8];
  float s = 0.f;
#pragma unroll
  for (int i = 0; i < 8; ++i) { vals[i] = row[lane + 32 * i]; s += vals[i]; }
#pragma unroll
  for (int off = 16; off > 0; off >>= 1) s += __shfl_xor(s, off, 32);
  float mu = s * (1.0f / 256.0f);
  float vs = 0.f;
#pragma unroll
  for (int i = 0; i < 8; ++i) { float d = vals[i] - mu; vs += d * d; }
#pragma unroll
  for (int off = 16; off > 0; off >>= 1) vs += __shfl_xor(vs, off, 32);
  float inv = rsqrtf(vs * (1.0f / 256.0f) + 1e-5f);
#pragma unroll
  for (int i = 0; i < 8; ++i) {
    int c = lane + 32 * i;
    y[(size_t)token * DDIM + c] = (vals[i] - mu) * inv * gamma[c] + beta[c];
  }
}

// sample embedding stage 1: t1[b,j] = relu(x[b,:] @ se_w1[:,j] + b1[j])
__global__ __launch_bounds__(256) void se1_kernel(
    const float* __restrict__ x, const float* __restrict__ w1,
    const float* __restrict__ b1, float* __restrict__ t1) {
  const int b = blockIdx.x, j = threadIdx.x;
  float s = b1[j];
  for (int g = 0; g < GG; ++g) s += x[(size_t)b * GG + g] * w1[(size_t)g * 256 + j];
  t1[b * 256 + j] = fmaxf(s, 0.f);
}

// sample embedding stage 2: samp[b,j] = t1[b,:] @ se_w2[:,j] + b2[j]
__global__ __launch_bounds__(256) void se2_kernel(
    const float* __restrict__ t1, const float* __restrict__ w2,
    const float* __restrict__ b2, float* __restrict__ samp) {
  const int b = blockIdx.x, j = threadIdx.x;
  float s = b2[j];
  for (int i = 0; i < 256; ++i) s += t1[b * 256 + i] * w2[(size_t)i * DDIM + j];
  samp[b * DDIM + j] = s;
}

// h0 = gene_emb + rotary-expression-embedding + samp
__global__ __launch_bounds__(256) void embed_kernel(
    const float* __restrict__ x, const float* __restrict__ gene_emb,
    const float* __restrict__ samp, float* __restrict__ h) {
  size_t idx = (size_t)blockIdx.x * 256 + threadIdx.x;
  if (idx >= (size_t)BB * GG * 128) return;
  int i = (int)(idx & 127);
  size_t bg = idx >> 7;
  int g = (int)(bg % GG);
  int b = (int)(bg / GG);
  float xv = x[(size_t)b * GG + g];
  float inv_freq = __expf(-(2.0f * i / 256.0f) * 4.60517018598809f);  // 100^-(2i/256)
  float f = xv * inv_freq;
  float sv = __sinf(f), cv = __cosf(f);
  if (xv == -10.0f) { sv = 0.f; cv = 0.f; }
  size_t base = ((size_t)b * GG + g) * DDIM;
  h[base + i]       = gene_emb[(size_t)g * DDIM + i]       + sv + samp[b * DDIM + i];
  h[base + 128 + i] = gene_emb[(size_t)g * DDIM + 128 + i] + cv + samp[b * DDIM + 128 + i];
}

// head: out[bg] = relu(t[bg,:] @ head_w2 + head_b2)
__global__ __launch_bounds__(256) void head_kernel(
    const float* __restrict__ t, const float* __restrict__ w2,
    const float* __restrict__ b2, float* __restrict__ out) {
  size_t token = (size_t)blockIdx.x * 256 + threadIdx.x;
  float s = b2[0];
  for (int j = 0; j < FF; ++j) s += t[token * FF + j] * w2[j];
  out[token] = fmaxf(s, 0.f);
}

// ---------------------------------------------------------------------------
extern "C" void kernel_launch(void* const* d_in, const int* in_sizes, int n_in,
                              void* d_out, int out_size, void* d_ws,
                              size_t ws_size, hipStream_t stream) {
  (void)in_sizes; (void)n_in; (void)out_size; (void)ws_size;
  const float* x         = (const float*)d_in[0];
  const float* gene_emb  = (const float*)d_in[1];
  const float* se_w1     = (const float*)d_in[2];
  const float* se_b1     = (const float*)d_in[3];
  const float* se_w2     = (const float*)d_in[4];
  const float* se_b2     = (const float*)d_in[5];
  const float* proj_w1   = (const float*)d_in[6];
  const float* proj_b1   = (const float*)d_in[7];
  const float* proj_w2   = (const float*)d_in[8];
  const float* proj_b2   = (const float*)d_in[9];
  const float* ln1_g     = (const float*)d_in[10];
  const float* ln1_b     = (const float*)d_in[11];
  const float* wq        = (const float*)d_in[12];
  const float* wk        = (const float*)d_in[13];
  const float* wv        = (const float*)d_in[14];
  const float* wo        = (const float*)d_in[15];
  const float* bo        = (const float*)d_in[16];
  const float* ln2_g     = (const float*)d_in[17];
  const float* ln2_b     = (const float*)d_in[18];
  const float* ff_w1     = (const float*)d_in[19];
  const float* ff_b1     = (const float*)d_in[20];
  const float* ff_w2     = (const float*)d_in[21];
  const float* ff_b2     = (const float*)d_in[22];
  const float* feat_proj = (const float*)d_in[23];
  const float* norm_g    = (const float*)d_in[24];
  const float* norm_b    = (const float*)d_in[25];
  const float* head_w1   = (const float*)d_in[26];
  const float* head_b1   = (const float*)d_in[27];
  const float* head_w2   = (const float*)d_in[28];
  const float* head_b2   = (const float*)d_in[29];
  float* out = (float*)d_out;

  char* wsb = (char*)d_ws;
  size_t off = 0;
  auto alloc = [&](size_t bytes) -> void* {
    void* p = wsb + off;
    off += (bytes + 255) & ~(size_t)255;
    return p;
  };
  const size_t fBGD = (size_t)BG * DDIM * sizeof(float);
  float* t1    = (float*)alloc((size_t)BB * 256 * 4);
  float* samp  = (float*)alloc((size_t)BB * DDIM * 4);
  float* hA    = (float*)alloc(fBGD);
  float* hB    = (float*)alloc(fBGD);
  float* aBuf  = (float*)alloc(fBGD);
  float* qBuf  = (float*)alloc(fBGD);
  float* kBuf  = (float*)alloc(fBGD);
  float* vBuf  = (float*)alloc(fBGD);
  float* yBuf  = (float*)alloc(fBGD);
  float* oBuf  = (float*)alloc(fBGD);
  float* ffBuf = (float*)alloc((size_t)BG * FF * 4);
  unsigned short* qp = (unsigned short*)alloc((size_t)BB * HH * GG * MFEA * 2);
  unsigned short* kp = (unsigned short*)alloc((size_t)BB * HH * GG * MFEA * 2);
  float* rowmax = (float*)alloc((size_t)BB * HH * GG * 4);
  float* kmax   = (float*)alloc((size_t)BB * HH * 4);
  float* ksum   = (float*)alloc((size_t)BB * HH * MFEA * 4);
  float* ctx    = (float*)alloc((size_t)BB * HH * MFEA * DH * 4);
  float* dinv   = (float*)alloc((size_t)BB * HH * GG * 4);

  dim3 blk(256);
  // sample embedding + token assembly
  se1_kernel<<<dim3(BB), blk, 0, stream>>>(x, se_w1, se_b1, t1);
  se2_kernel<<<dim3(BB), blk, 0, stream>>>(t1, se_w2, se_b2, samp);
  embed_kernel<<<dim3((unsigned)(((size_t)BB * GG * 128) / 256)), blk, 0, stream>>>(
      x, gene_emb, samp, hB);
  // projection MLP: hA = relu(h0 @ w1 + b1) @ w2 + b2
  gemm_bf16_wmma<1><<<dim3(BG / 64, FF / 128), blk, 0, stream>>>(
      hB, proj_w1, proj_b1, nullptr, nullptr, ffBuf, FF, DDIM);
  gemm_bf16_wmma<0><<<dim3(BG / 64, DDIM / 128), blk, 0, stream>>>(
      ffBuf, proj_w2, proj_b2, nullptr, nullptr, hA, DDIM, FF);

  float* hCur = hA;
  float* hAlt = hB;
  for (int l = 0; l < LLAY; ++l) {
    ln_kernel<<<dim3(BG / 8), blk, 0, stream>>>(hCur, ln1_g + l * DDIM,
                                                ln1_b + l * DDIM, aBuf);
    gemm_bf16_wmma<0><<<dim3(BG / 64, DDIM / 128), blk, 0, stream>>>(
        aBuf, wq + (size_t)l * DDIM * DDIM, nullptr, nullptr, nullptr, qBuf, DDIM, DDIM);
    gemm_bf16_wmma<0><<<dim3(BG / 64, DDIM / 128), blk, 0, stream>>>(
        aBuf, wk + (size_t)l * DDIM * DDIM, nullptr, nullptr, nullptr, kBuf, DDIM, DDIM);
    gemm_bf16_wmma<0><<<dim3(BG / 64, DDIM / 128), blk, 0, stream>>>(
        aBuf, wv + (size_t)l * DDIM * DDIM, nullptr, nullptr, nullptr, vBuf, DDIM, DDIM);
    const float* fp = feat_proj + (size_t)l * MFEA * DH;
    feat_kernel<0><<<dim3(GG / 16, BB * HH), blk, 0, stream>>>(qBuf, fp, qp, rowmax, kmax);
    feat_kernel<1><<<dim3(GG / 16, BB * HH), blk, 0, stream>>>(kBuf, fp, kp, rowmax, kmax);
    reduce_max_kernel<<<dim3(BB * HH), blk, 0, stream>>>(rowmax, kmax);
    feat_kernel<2><<<dim3(GG / 16, BB * HH), blk, 0, stream>>>(kBuf, fp, kp, rowmax, kmax);
    ksum_kernel<<<dim3(BB * HH), dim3(128), 0, stream>>>(kp, ksum);
    ctx_kernel<<<dim3(BB * HH), blk, 0, stream>>>(kp, vBuf, ctx);
    dinv_kernel<<<dim3((GG + 255) / 256, BB * HH), blk, 0, stream>>>(qp, ksum, dinv);
    attnout_kernel<<<dim3((GG + 63) / 64, BB * HH), blk, 0, stream>>>(qp, ctx, dinv, oBuf);
    gemm_bf16_wmma<0><<<dim3(BG / 64, DDIM / 128), blk, 0, stream>>>(
        oBuf, wo + (size_t)l * DDIM * DDIM, bo + l * DDIM, hCur, nullptr, yBuf, DDIM, DDIM);
    ln_kernel<<<dim3(BG / 8), blk, 0, stream>>>(yBuf, ln2_g + l * DDIM,
                                                ln2_b + l * DDIM, aBuf);
    gemm_bf16_wmma<2><<<dim3(BG / 64, FF / 128), blk, 0, stream>>>(
        aBuf, ff_w1 + (size_t)l * DDIM * FF, ff_b1 + l * FF, nullptr, nullptr,
        ffBuf, FF, DDIM);
    gemm_bf16_wmma<0><<<dim3(BG / 64, DDIM / 128), blk, 0, stream>>>(
        ffBuf, ff_w2 + (size_t)l * FF * DDIM, ff_b2 + l * DDIM, yBuf, hCur,
        hAlt, DDIM, FF);
    float* tmp = hCur; hCur = hAlt; hAlt = tmp;
  }
  ln_kernel<<<dim3(BG / 8), blk, 0, stream>>>(hCur, norm_g, norm_b, aBuf);
  gemm_bf16_wmma<1><<<dim3(BG / 64, FF / 128), blk, 0, stream>>>(
      aBuf, head_w1, head_b1, nullptr, nullptr, ffBuf, FF, DDIM);
  head_kernel<<<dim3(BG / 256), blk, 0, stream>>>(ffBuf, head_w2, head_b2, out);
}
